// ConfidenceMaskedDecoder_3590592659665
// MI455X (gfx1250) — matrix operand order repaired and checked by reading, not compiled, not run on hardware
//
#include <hip/hip_runtime.h>
#include <hip/hip_bf16.h>
#include <math.h>

// Problem constants (from reference)
#define BB 2
#define SS 2048
#define VV 32000
#define DD 1024
#define HH 512
#define MM (BB * SS)   // 4096 tokens

typedef __attribute__((ext_vector_type(16))) __bf16 bf16x16;
typedef __attribute__((ext_vector_type(8)))  __bf16 bf16x8;
typedef __attribute__((ext_vector_type(8)))  float  floatx8;
typedef __attribute__((ext_vector_type(4)))  float  fx4;

// ---------------------------------------------------------------------------
// Kernel 1: softmax stats over V=32000 per token (the ~0.5 GB dominant stream).
// One block (1024 threads) per token; each lane holds 8 float4 (32 logits) in
// registers => exactly one HBM pass. Loads are b128 + non-temporal (stream-
// once data must not evict the L2 working set used by later kernels).
// maxp = 1/Z (softmax max prob), entropy uses exact log(p + 1e-8) with the
// hardware v_log_f32 (__logf) so the kernel stays memory-bound.
// ---------------------------------------------------------------------------
__global__ void __launch_bounds__(1024)
conf_logits_kernel(const float* __restrict__ logits,
                   float* __restrict__ maxp_out,
                   float* __restrict__ entc_out)
{
    const int token = blockIdx.x;
    const int t     = threadIdx.x;           // 0..1023
    const fx4* row  = (const fx4*)(logits + (size_t)token * VV);  // 8000 fx4

    __shared__ float red[32];

    fx4 v[8];
    #pragma unroll
    for (int i = 0; i < 8; ++i) {
        int idx = t + i * 1024;              // fx4 index; tail: 8000..8191 invalid
        if (idx < (VV / 4)) {
            v[i] = __builtin_nontemporal_load(&row[idx]);
        } else {
            fx4 ninf; ninf[0] = ninf[1] = ninf[2] = ninf[3] = -INFINITY;
            v[i] = ninf;
        }
    }

    // ---- block max ----
    float m = -INFINITY;
    #pragma unroll
    for (int i = 0; i < 8; ++i)
        #pragma unroll
        for (int j = 0; j < 4; ++j) m = fmaxf(m, v[i][j]);
    #pragma unroll
    for (int o = 16; o > 0; o >>= 1) m = fmaxf(m, __shfl_xor(m, o, 32));
    if ((t & 31) == 0) red[t >> 5] = m;
    __syncthreads();
    float M = -INFINITY;
    #pragma unroll
    for (int i = 0; i < 32; ++i) M = fmaxf(M, red[i]);
    __syncthreads();

    // ---- block sum of exp ----
    fx4 e[8];
    float se = 0.0f;
    #pragma unroll
    for (int i = 0; i < 8; ++i)
        #pragma unroll
        for (int j = 0; j < 4; ++j) { e[i][j] = __expf(v[i][j] - M); se += e[i][j]; }
    #pragma unroll
    for (int o = 16; o > 0; o >>= 1) se += __shfl_xor(se, o, 32);
    if ((t & 31) == 0) red[t >> 5] = se;
    __syncthreads();
    float Z = 0.0f;
    #pragma unroll
    for (int i = 0; i < 32; ++i) Z += red[i];
    __syncthreads();

    // ---- entropy: -sum p * log(p + 1e-8), p = e/Z ----
    const float invZ = 1.0f / Z;
    float hs = 0.0f;
    #pragma unroll
    for (int i = 0; i < 8; ++i)
        #pragma unroll
        for (int j = 0; j < 4; ++j) {
            float p = e[i][j] * invZ;
            hs += p * __logf(p + 1e-8f);     // p==0 contributes 0
        }
    #pragma unroll
    for (int o = 16; o > 0; o >>= 1) hs += __shfl_xor(hs, o, 32);
    if ((t & 31) == 0) red[t >> 5] = hs;
    __syncthreads();
    float Hsum = 0.0f;
    #pragma unroll
    for (int i = 0; i < 32; ++i) Hsum += red[i];

    if (t == 0) {
        maxp_out[token] = invZ;                              // exp(M-M)/Z
        entc_out[token] = 1.0f + Hsum * (1.0f / 10.373491f); // 1 - (-Hsum)/ln(32000)
    }
}

// ---------------------------------------------------------------------------
// Kernel 2: per-token prep over hidden_states [4096, 1024]:
//   - convert row to bf16 (A matrix for WMMA GEMM)
//   - L2 norm of the row
//   - raw adjacent dot h_i . h_{i+1} (within same batch)
// ---------------------------------------------------------------------------
__global__ void __launch_bounds__(256)
prep_hidden_kernel(const float* __restrict__ hs,
                   __bf16* __restrict__ Abf,
                   float* __restrict__ norms,
                   float* __restrict__ adjdot)
{
    const int i = blockIdx.x;            // token
    const int s = i % SS;
    const int t = threadIdx.x;           // 0..255

    __shared__ float red[8];
    __shared__ float red2[8];

    const fx4 h = ((const fx4*)(hs + (size_t)i * DD))[t];

    __bf16* op = Abf + (size_t)i * DD + t * 4;
    op[0] = (__bf16)h[0]; op[1] = (__bf16)h[1]; op[2] = (__bf16)h[2]; op[3] = (__bf16)h[3];

    float ss = h[0]*h[0] + h[1]*h[1] + h[2]*h[2] + h[3]*h[3];

    float dt = 0.0f;
    if (s < SS - 1) {
        const fx4 g = ((const fx4*)(hs + (size_t)(i + 1) * DD))[t];
        dt = h[0]*g[0] + h[1]*g[1] + h[2]*g[2] + h[3]*g[3];
    }

    #pragma unroll
    for (int o = 16; o > 0; o >>= 1) {
        ss += __shfl_xor(ss, o, 32);
        dt += __shfl_xor(dt, o, 32);
    }
    if ((t & 31) == 0) { red[t >> 5] = ss; red2[t >> 5] = dt; }
    __syncthreads();
    if (t == 0) {
        float sst = 0.0f, dtt = 0.0f;
        #pragma unroll
        for (int w = 0; w < 8; ++w) { sst += red[w]; dtt += red2[w]; }
        norms[i]  = fmaxf(sqrtf(sst), 1e-12f);
        adjdot[i] = (s < SS - 1) ? dtt : 0.0f;
    }
}

// ---------------------------------------------------------------------------
// Kernel 3: W1 [D=1024, H=512] f32  ->  W1T bf16 [H][D]  (B operand, K-contig)
// ---------------------------------------------------------------------------
__global__ void __launch_bounds__(256)
prep_w1t_kernel(const float* __restrict__ W1, __bf16* __restrict__ W1T)
{
    const int n = blockIdx.x;            // 0..511
    const int t = threadIdx.x;
    #pragma unroll
    for (int j = 0; j < 4; ++j) {
        int k = t + j * 256;
        W1T[(size_t)n * DD + k] = (__bf16)W1[(size_t)k * HH + n];
    }
}

// ---------------------------------------------------------------------------
// Kernel 4: learned head via WMMA bf16 GEMM, fused GELU/W2/sigmoid epilogue.
// Workgroup = 16 tokens (M-tile). A tile (16 x 1024 bf16 = 32 KB) staged into
// LDS with the CDNA5 async DMA path (global_load_async_to_lds_b128 +
// s_wait_asynccnt). 8 waves x 4 passes cover N=512 in 16-wide tiles; K loop
// issues 32 v_wmma_f32_16x16x32_bf16 per N-tile. Epilogue reduces over N with
// shuffles and ds_add_f32 into per-token LDS accumulators.
// ---------------------------------------------------------------------------
__global__ void __launch_bounds__(256)
head_gemm_kernel(const __bf16* __restrict__ Abf,
                 const __bf16* __restrict__ W1T,
                 const float*  __restrict__ b1,
                 const float*  __restrict__ W2,
                 const float*  __restrict__ b2,
                 float* __restrict__ learned)
{
    __shared__ __bf16 sA[16 * DD];       // 32 KB
    __shared__ float  accum[16];

    const int t      = threadIdx.x;      // 0..255
    const int token0 = blockIdx.x * 16;

    // Stage A tile via async DMA: 2048 x b128 chunks, 8 per thread.
    {
        const __bf16* gbase = Abf + (size_t)token0 * DD;
        #pragma unroll
        for (int j = 0; j < 8; ++j) {
            const int idx = (t + j * 256) * 8;                  // bf16 element index
            const unsigned lds_off = (unsigned)(uintptr_t)&sA[idx];
            const void* gptr = (const void*)(gbase + idx);
            asm volatile("global_load_async_to_lds_b128 %0, %1, off"
                         :: "v"(lds_off), "v"(gptr) : "memory");
        }
    }
    if (t < 16) accum[t] = 0.0f;
    asm volatile("s_wait_asynccnt 0" ::: "memory");
    __syncthreads();

    const int wave = t >> 5;
    const int lane = t & 31;
    const int half = lane >> 4;          // 0 or 1
    const int mlo  = lane & 15;

    for (int pass = 0; pass < 4; ++pass) {
        const int n  = (wave + pass * 8) * 16 + mlo;    // this lane's N column
        const __bf16* Brow = W1T + (size_t)n * DD;

        floatx8 acc = {0.f, 0.f, 0.f, 0.f, 0.f, 0.f, 0.f, 0.f};

        for (int kk = 0; kk < DD; kk += 32) {
            // A fragment (16-bit 16x32 layout): lanes 0-15 -> K=kk+0..7 & kk+16..23,
            // lanes 16-31 -> K=kk+8..15 & kk+24..31, row M = lane&15.
            const int abase = kk + half * 8;
            const bf16x8 alo = *(const bf16x8*)&sA[mlo * DD + abase];
            const bf16x8 ahi = *(const bf16x8*)&sA[mlo * DD + abase + 16];
            // B fragment (16-bit 32x16): lanes 0-15 -> K=kk+0..7 & kk+16..23,
            // lanes 16-31 -> K=kk+8..15 & kk+24..31, column N = lane&15.
            const int kb = kk + half * 8;
            const bf16x8 blo = *(const bf16x8*)&Brow[kb];
            const bf16x8 bhi = *(const bf16x8*)&Brow[kb + 16];

            __builtin_prefetch(&Brow[kb + 32], 0, 3);   // global_prefetch_b8 (near)

            bf16x16 afrag, bfrag;
            #pragma unroll
            for (int j = 0; j < 8; ++j) {
                afrag[j] = alo[j]; afrag[j + 8] = ahi[j];
                bfrag[j] = blo[j]; bfrag[j + 8] = bhi[j];
            }
            acc = __builtin_amdgcn_wmma_f32_16x16x32_bf16(
                      false, afrag, false, bfrag, (short)0, acc, false, false);
        }

        // Epilogue: lane holds (m = r + 8*half, n) for r=0..7.
        const float w2n = W2[n];
        const float bn  = b1[n];
        float part[8];
        #pragma unroll
        for (int r = 0; r < 8; ++r) {
            float x = acc[r] + bn;
            float g = 0.5f * x * (1.0f + erff(x * 0.70710678118f));   // exact GELU
            part[r] = g * w2n;
        }
        // Reduce over the 16 N-lanes within each half (xor bits 0..3 stay in half)
        #pragma unroll
        for (int o = 8; o > 0; o >>= 1) {
            #pragma unroll
            for (int r = 0; r < 8; ++r) part[r] += __shfl_xor(part[r], o, 32);
        }
        if (mlo == 0) {
            const int mbase = half * 8;
            #pragma unroll
            for (int r = 0; r < 8; ++r) atomicAdd(&accum[mbase + r], part[r]);
        }
    }
    __syncthreads();
    if (t < 16) {
        float x = accum[t] + b2[0];
        learned[token0 + t] = 1.0f / (1.0f + __expf(-x));
    }
}

// ---------------------------------------------------------------------------
// Kernel 5: final combine with context boost + mask.
// ---------------------------------------------------------------------------
__global__ void __launch_bounds__(256)
combine_kernel(const float* __restrict__ maxp,
               const float* __restrict__ entc,
               const float* __restrict__ learned,
               const float* __restrict__ norms,
               const float* __restrict__ adjdot,
               const unsigned char* __restrict__ mask,   // jnp bool_ = 1 byte
               float* __restrict__ out)
{
    const int i = blockIdx.x * blockDim.x + threadIdx.x;
    if (i >= MM) return;
    const int s = i % SS;

    const float ni = norms[i];
    float left = 0.0f, right = 0.0f;
    if (s > 0)      left  = adjdot[i - 1] / (norms[i - 1] * ni);
    if (s < SS - 1) right = adjdot[i]     / (ni * norms[i + 1]);
    const float nn  = (s > 0 && s < SS - 1) ? 2.0f : 1.0f;
    const float ctx = (left + right) / nn;
    const float boost = 1.0f / (1.0f + __expf(-2.0f * ctx));

    const float val = 0.4f * maxp[i] + 0.2f * entc[i]
                    + 0.2f * learned[i] + 0.2f * boost;
    out[i] = mask[i] ? val : 0.0f;
}

// ---------------------------------------------------------------------------
extern "C" void kernel_launch(void* const* d_in, const int* in_sizes, int n_in,
                              void* d_out, int out_size, void* d_ws, size_t ws_size,
                              hipStream_t stream) {
    const float* logits = (const float*)d_in[0];
    const float* hidden = (const float*)d_in[1];
    const unsigned char* mask = (const unsigned char*)d_in[2];
    const float* W1 = (const float*)d_in[3];
    const float* b1 = (const float*)d_in[4];
    const float* W2 = (const float*)d_in[5];
    const float* b2 = (const float*)d_in[6];
    float* out = (float*)d_out;

    // Workspace layout (256-byte aligned slabs)
    char* ws = (char*)d_ws;
    size_t off = 0;
    __bf16* Abf    = (__bf16*)(ws + off); off += (size_t)MM * DD * 2;   // 8 MB
    __bf16* W1T    = (__bf16*)(ws + off); off += (size_t)HH * DD * 2;   // 1 MB
    float*  norms  = (float*)(ws + off);  off += (size_t)MM * 4;
    float*  adjdot = (float*)(ws + off);  off += (size_t)MM * 4;
    float*  maxp   = (float*)(ws + off);  off += (size_t)MM * 4;
    float*  entc   = (float*)(ws + off);  off += (size_t)MM * 4;
    float*  learned= (float*)(ws + off);  off += (size_t)MM * 4;
    (void)ws_size; (void)in_sizes; (void)n_in; (void)out_size;

    conf_logits_kernel<<<MM, 1024, 0, stream>>>(logits, maxp, entc);
    prep_hidden_kernel<<<MM, 256, 0, stream>>>(hidden, Abf, norms, adjdot);
    prep_w1t_kernel<<<HH, 256, 0, stream>>>(W1, W1T);
    head_gemm_kernel<<<MM / 16, 256, 0, stream>>>(Abf, W1T, b1, W2, b2, learned);
    combine_kernel<<<(MM + 255) / 256, 256, 0, stream>>>(maxp, entc, learned,
                                                         norms, adjdot, mask, out);
}
